// VoxelFeatureExtractor_29712583754074
// MI455X (gfx1250) — matrix operand order repaired and checked by reading, not compiled, not run on hardware
//
#include <hip/hip_runtime.h>
#include <hip/hip_bf16.h>

// ---------------------------------------------------------------------------
// CDNA5 (gfx1250) implementation: every Linear layer runs on
// v_wmma_f32_16x16x32_f16 (f16 inputs, f32 accumulate). LayerNorm reduction
// is a pure-VALU DPP16 butterfly over the 16-lane half-wave (no LDS, no
// waits). Mish uses the closed form y*(t^2+2t)/(t^2+2t+2), t=e^y: one
// v_exp_f32 + one v_rcp_f32 per element. Intermediates stored as f16;
// the final scatter is inverted into one coalesced gather-write of the
// 268MB output via a 4MB cell->row inverse map (value buffers sit in L2).
// ---------------------------------------------------------------------------

typedef __attribute__((ext_vector_type(16))) _Float16 v16h;
typedef __attribute__((ext_vector_type(8)))  _Float16 v8h;
typedef __attribute__((ext_vector_type(8)))  float    v8f;

__device__ __forceinline__ v16h cat16(v8h lo, v8h hi) {
    return __builtin_shufflevector(lo, hi, 0,1,2,3,4,5,6,7,8,9,10,11,12,13,14,15);
}
__device__ __forceinline__ v8f zero8() {
    v8f r;
#pragma unroll
    for (int i = 0; i < 8; ++i) r[i] = 0.0f;
    return r;
}

// ---- DPP16 butterfly add: full sum over each 16-lane row, all in VALU -----
template<int CTRL>
__device__ __forceinline__ float dpp_add(float x) {
    int v = __builtin_amdgcn_update_dpp(0, __float_as_int(x), CTRL, 0xF, 0xF, true);
    return x + __int_as_float(v);
}
__device__ __forceinline__ float red16(float x) {
    x = dpp_add<0xB1>(x);    // quad_perm [1,0,3,2]  : lane ^ 1
    x = dpp_add<0x4E>(x);    // quad_perm [2,3,0,1]  : lane ^ 2
    x = dpp_add<0x141>(x);   // row_half_mirror      : lane ^ 7  -> quad ^ 1
    x = dpp_add<0x140>(x);   // row_mirror           : lane ^ 15 -> quad ^ 3
    return x;                // every lane now holds the 16-lane sum
}

// ---- fast Mish: y * tanh(softplus(y)) = y*(t^2+2t)/(t^2+2t+2), t=e^y ------
__device__ __forceinline__ float mish(float y) {
    float t = __expf(y);
    float u = t * (t + 2.0f);
    float m = y * u * __builtin_amdgcn_rcpf(u + 2.0f);
    return (y > 20.0f) ? y : m;   // t^2 stays finite for y<=20
}

// ---------------------------------------------------------------------------
// Weight prep: f32 [N][K] -> f16 [N][KP] (zero-padded K), packed in workspace.
// ---------------------------------------------------------------------------
template<int K, int KP, int N>
__global__ void prep_w(const float* __restrict__ W, _Float16* __restrict__ dst) {
    int i = blockIdx.x * 256 + threadIdx.x;
    if (i >= N * KP) return;
    int n = i / KP, k = i - n * KP;
    dst[i] = (k < K) ? (_Float16)W[n * K + k] : (_Float16)0.0f;
}

// ---------------------------------------------------------------------------
// Generic FE layer: Y = Mish(LayerNorm(X @ W^T))
//   one wave per 16-row tile; NT = N/16 accumulators; KB = KP/32 k-steps.
//   G==0 : dense rows from src (f32 if SRC_F32 with K<=32, else f16, stride K)
//   G>=1 : row i is concat of G gathered 32-wide f16 rows concv[gidx[i*G+g]]
// ---------------------------------------------------------------------------
template<int K, int KP, int N, int G, bool SRC_F32, bool OUT_F16, int WAVES>
__global__ __launch_bounds__(WAVES * 32)
void fe_kernel(const void* __restrict__ src, const int* __restrict__ gidx,
               const _Float16* __restrict__ wp,
               const float* __restrict__ gamma, const float* __restrict__ beta,
               void* __restrict__ dst, int M)
{
    constexpr int KB = KP / 32;
    constexpr int NT = N / 16;
    const int lane = threadIdx.x & 31;
    const int wave = threadIdx.x >> 5;
    const int hi   = lane >> 4;       // half-wave select
    const int ln   = lane & 15;
    const int tile = blockIdx.x * WAVES + wave;
    const int row0 = tile * 16;
    if (row0 >= M) return;            // wave-uniform; EXEC stays all-ones
    const int arow = row0 + ln;       // A-matrix row this lane feeds

    v8f acc[NT];
#pragma unroll
    for (int nt = 0; nt < NT; ++nt) acc[nt] = zero8();

#pragma unroll
    for (int kb = 0; kb < KB; ++kb) {
        // ---- A fragment (16x32 f16): lane ln holds K = klo..klo+7, klo+16..klo+23
        v16h a;
        if constexpr (G > 0) {
            const int srow = gidx[arow * G + kb];
            const _Float16* p = (const _Float16*)src + (size_t)srow * 32 + hi * 8;
            a = cat16(*(const v8h*)p, *(const v8h*)(p + 16));
        } else if constexpr (SRC_F32) {
            const float* rowf = (const float*)src + (size_t)arow * K;
#pragma unroll
            for (int e = 0; e < 16; ++e) {
                const int k = kb * 32 + (e & 7) + ((e >> 3) << 4) + hi * 8;
                a[e] = (k < K) ? (_Float16)rowf[k] : (_Float16)0.0f;
            }
        } else {
            const _Float16* p = (const _Float16*)src + (size_t)arow * K + kb * 32 + hi * 8;
            a = cat16(*(const v8h*)p, *(const v8h*)(p + 16));
        }
        // ---- B fragments (32x16 f16): lane col = ln, K = klo..klo+7, klo+16..klo+23
#pragma unroll
        for (int nt = 0; nt < NT; ++nt) {
            const _Float16* q = wp + (size_t)(nt * 16 + ln) * KP + kb * 32 + hi * 8;
            v16h b = cat16(*(const v8h*)q, *(const v8h*)(q + 16));
            acc[nt] = __builtin_amdgcn_wmma_f32_16x16x32_f16(
                false, a, false, b, (short)0, acc[nt], false, false);
        }
    }

    // ---- LayerNorm over N + Mish.  D layout: VGPR r, lane -> row r+8*hi, col ln.
    float gam[NT], bet[NT];
#pragma unroll
    for (int nt = 0; nt < NT; ++nt) { gam[nt] = gamma[nt * 16 + ln]; bet[nt] = beta[nt * 16 + ln]; }

    const float invN = 1.0f / (float)N;
#pragma unroll
    for (int r = 0; r < 8; ++r) {
        float s = 0.0f, p = 0.0f;
#pragma unroll
        for (int nt = 0; nt < NT; ++nt) { float y = acc[nt][r]; s += y; p += y * y; }
        s = red16(s);                 // two independent DPP chains -> good ILP
        p = red16(p);
        const float mu  = s * invN;
        const float var = fmaxf(p * invN - mu * mu, 0.0f);
        const float inv = rsqrtf(var + 1e-5f);
        const size_t orow = (size_t)(row0 + r + hi * 8);
#pragma unroll
        for (int nt = 0; nt < NT; ++nt) {
            float y = (acc[nt][r] - mu) * inv * gam[nt] + bet[nt];
            float o = mish(y);
            if constexpr (OUT_F16)
                ((_Float16*)dst)[orow * N + nt * 16 + ln] = (_Float16)o;
            else
                ((float*)dst)[orow * N + nt * 16 + ln] = o;
        }
    }
}

// ---------------------------------------------------------------------------
// Inverse scatter map + single coalesced BEV emit (out is [C=64][W=1024][H=1024])
// ---------------------------------------------------------------------------
__global__ void map_init(int* __restrict__ map) {
    int i = blockIdx.x * 256 + threadIdx.x;
    if (i < 1024 * 1024) map[i] = 0;
}
__global__ void map_fill(const int* __restrict__ coors, int* __restrict__ map, int n, int tag) {
    int i = blockIdx.x * 256 + threadIdx.x;
    if (i >= n) return;
    int h = coors[2 * i], w = coors[2 * i + 1];
    map[(w << 10) | h] = (tag << 24) | i;
}
__global__ void bev_emit(const int* __restrict__ map,
                         const float* __restrict__ poorb,
                         const float* __restrict__ normb,
                         const float* __restrict__ richb,
                         float* __restrict__ out) {
    const size_t t    = (size_t)blockIdx.x * 256 + threadIdx.x;   // 16,777,216 threads
    const size_t base = t * 4;
    const int c   = (int)(base >> 20);
    const int wh0 = (int)(base & 0xFFFFF);
    float vals[4];
#pragma unroll
    for (int j = 0; j < 4; ++j) {
        int m = map[wh0 + j];
        float v = 0.0f;
        if (m) {
            int tag = m >> 24, i = m & 0xFFFFFF;
            if (tag == 1)      { if (c < 16) v = poorb[(size_t)i * 16 + c]; }
            else if (tag == 2) { if (c < 32) v = normb[(size_t)i * 32 + c]; }
            else               {              v = richb[(size_t)i * 64 + c]; }
        }
        vals[j] = v;
    }
    *(float4*)(out + base) = make_float4(vals[0], vals[1], vals[2], vals[3]);
}

// ---------------------------------------------------------------------------
// Workspace layout (bytes); all 256B aligned.
// ---------------------------------------------------------------------------
static constexpr size_t WP_PV1 = 0;       // 32x32   f16
static constexpr size_t WP_NV1 = 1024;    // 32x32
static constexpr size_t WP_NV2 = 2048;    // 32x256
static constexpr size_t WP_PB1 = 10240;   // 16x64
static constexpr size_t WP_NB1 = 11264;   // 32x32
static constexpr size_t WP_NB2 = 12288;   // 32x256
static constexpr size_t WP_RB1 = 20480;   // 64x32
static constexpr size_t WP_RB2 = 22528;   // 64x64
static constexpr size_t WP_RB3 = 26624;   // 64x1024
static constexpr size_t WP_RB4 = 92160;   // 64x64   (pool ends at 96256 halves)

static constexpr size_t OFF_POORB = 262144;                    // 300000*16*4
static constexpr size_t OFF_NORMB = OFF_POORB + 19200000;      // 200000*32*4
static constexpr size_t OFF_RICHB = OFF_NORMB + 25600000;      // 100000*64*4
static constexpr size_t OFF_MAP   = OFF_RICHB + 25600000;      // 1M * 4
static constexpr size_t OFF_CONCV = OFF_MAP   + 4194304;       // 700000*32*2 f16
static constexpr size_t OFF_X     = OFF_CONCV + 44800000;      // 204.8MB reuse region
static constexpr size_t OFF_Y     = OFF_X     + 204800000;     // 204.8MB

extern "C" void kernel_launch(void* const* d_in, const int* in_sizes, int n_in,
                              void* d_out, int out_size, void* d_ws, size_t ws_size,
                              hipStream_t stream)
{
    (void)n_in; (void)out_size; (void)ws_size;
    const float* poorvoxels = (const float*)d_in[0];
    const float* normvoxels = (const float*)d_in[1];

    struct P { const float* W; const float* g; const float* b; };
    P pv1, nv1, nv2, pb1, nb1, nb2, rb1, rb2, rb3, rb4;
    const int pbase = 2;
    auto leaf = [&](int i) { return (const float*)d_in[pbase + i]; };
    // Detect flattening order of the nested params dict:
    //   insertion order -> first leaf is pv1.W (448 elems), leaves are (W,g,b)
    //   jax sorted keys -> first leaf is nb1.W (1024 elems), leaves are (W,b,g)
    if (in_sizes[pbase] == 448) {
        auto T = [&](int j) { return P{ leaf(3 * j + 0), leaf(3 * j + 1), leaf(3 * j + 2) }; };
        pv1 = T(0); nv1 = T(1); nv2 = T(2); pb1 = T(3); nb1 = T(4);
        nb2 = T(5); rb1 = T(6); rb2 = T(7); rb3 = T(8); rb4 = T(9);
    } else {
        auto T = [&](int j) { return P{ leaf(3 * j + 0), leaf(3 * j + 2), leaf(3 * j + 1) }; };
        nb1 = T(0); nb2 = T(1); nv1 = T(2); nv2 = T(3); pb1 = T(4);
        pv1 = T(5); rb1 = T(6); rb2 = T(7); rb3 = T(8); rb4 = T(9);
    }
    const int* poorbevsidx = (const int*)d_in[32];
    const int* normbevsidx = (const int*)d_in[33];
    const int* richbevsidx = (const int*)d_in[34];
    const int* poorcoors   = (const int*)d_in[35];
    const int* normcoors   = (const int*)d_in[36];
    const int* richcoors   = (const int*)d_in[37];

    char* ws = (char*)d_ws;
    _Float16* wpool  = (_Float16*)ws;
    float*    poorb  = (float*)(ws + OFF_POORB);
    float*    normb  = (float*)(ws + OFF_NORMB);
    float*    richb  = (float*)(ws + OFF_RICHB);
    int*      cellmap= (int*)  (ws + OFF_MAP);
    _Float16* concv  = (_Float16*)(ws + OFF_CONCV);
    _Float16* bufX   = (_Float16*)(ws + OFF_X);
    _Float16* bufY   = (_Float16*)(ws + OFF_Y);

    // ---- pack all weights as padded f16 (tiny; stays L2 resident)
    prep_w<14,  32, 32><<<   4, 256, 0, stream>>>(pv1.W, wpool + WP_PV1);
    prep_w< 7,  32, 32><<<   4, 256, 0, stream>>>(nv1.W, wpool + WP_NV1);
    prep_w<256,256, 32><<<  32, 256, 0, stream>>>(nv2.W, wpool + WP_NV2);
    prep_w<64,  64, 16><<<   4, 256, 0, stream>>>(pb1.W, wpool + WP_PB1);
    prep_w<32,  32, 32><<<   4, 256, 0, stream>>>(nb1.W, wpool + WP_NB1);
    prep_w<256,256, 32><<<  32, 256, 0, stream>>>(nb2.W, wpool + WP_NB2);
    prep_w<32,  32, 64><<<   8, 256, 0, stream>>>(rb1.W, wpool + WP_RB1);
    prep_w<64,  64, 64><<<  16, 256, 0, stream>>>(rb2.W, wpool + WP_RB2);
    prep_w<1024,1024,64><<<256, 256, 0, stream>>>(rb3.W, wpool + WP_RB3);
    prep_w<64,  64, 64><<<  16, 256, 0, stream>>>(rb4.W, wpool + WP_RB4);

    constexpr int WAVES = 8;
    auto nb = [](int M) { return (M + 16 * WAVES - 1) / (16 * WAVES); };

    // ---- voxel FEs -> concv (f16 [700000][32])
    fe_kernel<14, 32, 32, 0, true,  true,  WAVES><<<nb( 400000), 256, 0, stream>>>(
        poorvoxels, nullptr, wpool + WP_PV1, pv1.g, pv1.b, concv, 400000);
    fe_kernel< 7, 32, 32, 0, true,  true,  WAVES><<<nb(2400000), 256, 0, stream>>>(
        normvoxels, nullptr, wpool + WP_NV1, nv1.g, nv1.b, bufX, 2400000);
    fe_kernel<256,256,32, 0, false, true,  WAVES><<<nb( 300000), 256, 0, stream>>>(
        bufX, nullptr, wpool + WP_NV2, nv2.g, nv2.b, concv + (size_t)400000 * 32, 300000);

    // ---- BEV-cell FEs (gathers from concv)
    fe_kernel<64, 64, 16, 2, false, false, WAVES><<<nb( 300000), 256, 0, stream>>>(
        concv, poorbevsidx, wpool + WP_PB1, pb1.g, pb1.b, poorb, 300000);
    fe_kernel<32, 32, 32, 1, false, true,  WAVES><<<nb(1600000), 256, 0, stream>>>(
        concv, normbevsidx, wpool + WP_NB1, nb1.g, nb1.b, bufX, 1600000);
    fe_kernel<256,256,32, 0, false, false, WAVES><<<nb( 200000), 256, 0, stream>>>(
        bufX, nullptr, wpool + WP_NB2, nb2.g, nb2.b, normb, 200000);
    fe_kernel<32, 32, 64, 1, false, true,  WAVES><<<nb(1600000), 256, 0, stream>>>(
        concv, richbevsidx, wpool + WP_RB1, rb1.g, rb1.b, bufX, 1600000);
    fe_kernel<64, 64, 64, 0, false, true,  WAVES><<<nb(1600000), 256, 0, stream>>>(
        bufX, nullptr, wpool + WP_RB2, rb2.g, rb2.b, bufY, 1600000);
    fe_kernel<1024,1024,64,0,false, true,  WAVES><<<nb( 100000), 256, 0, stream>>>(
        bufY, nullptr, wpool + WP_RB3, rb3.g, rb3.b, bufX, 100000);
    fe_kernel<64, 64, 64, 0, false, false, WAVES><<<nb( 100000), 256, 0, stream>>>(
        bufX, nullptr, wpool + WP_RB4, rb4.g, rb4.b, richb, 100000);

    // ---- inverse map + single coalesced output write
    map_init<<<4096, 256, 0, stream>>>(cellmap);
    map_fill<<<(300000 + 255) / 256, 256, 0, stream>>>(poorcoors, cellmap, 300000, 1);
    map_fill<<<(200000 + 255) / 256, 256, 0, stream>>>(normcoors, cellmap, 200000, 2);
    map_fill<<<(100000 + 255) / 256, 256, 0, stream>>>(richcoors, cellmap, 100000, 3);
    bev_emit<<<65536, 256, 0, stream>>>(cellmap, poorb, normb, richb, (float*)d_out);
}
